// MolecularGCN_79276506349573
// MI455X (gfx1250) — matrix operand order repaired and checked by reading, not compiled, not run on hardware
//
#include <hip/hip_runtime.h>
#include <hip/hip_bf16.h>
#include <math.h>

typedef __attribute__((ext_vector_type(16))) _Float16 v16h;
typedef __attribute__((ext_vector_type(8)))  _Float16 v8h;
typedef __attribute__((ext_vector_type(8)))  float    v8f;

#define NNODES 10000
#define NEDGES 160000
#define NGRAPH 128
#define NHEADS 4
#define DMAX   512
#define OUTSTRIDE 2048

// ---------------------------------------------------------------- utilities

__device__ __forceinline__ float leaky02(float v) { return v > 0.f ? v : 0.2f * v; }

// float atomic max via int/uint ordering trick (works for mixed signs)
__device__ __forceinline__ void atomicMaxF(float* addr, float v) {
  if (v >= 0.0f) atomicMax((int*)addr, __float_as_int(v));
  else           atomicMin((unsigned int*)addr, __float_as_uint(v));
}

__global__ void fill_kernel(float* __restrict__ p, float v, long long n) {
  long long i = (long long)blockIdx.x * blockDim.x + threadIdx.x;
  long long s = (long long)gridDim.x * blockDim.x;
  for (; i < n; i += s) p[i] = v;
}

// f32 [M x K] -> f16 [M x Kp], zero-padded columns (A operands)
__global__ void cvt_a_kernel(const float* __restrict__ src, _Float16* __restrict__ dst,
                             int M, int K, int Kp) {
  long long total = (long long)M * Kp;
  long long i = (long long)blockIdx.x * blockDim.x + threadIdx.x;
  long long s = (long long)gridDim.x * blockDim.x;
  for (; i < total; i += s) {
    int m = (int)(i / Kp), k = (int)(i % Kp);
    dst[i] = (k < K) ? (_Float16)src[(long long)m * K + k] : (_Float16)0.f;
  }
}

// f32 [K x N] -> f16 [Kp x N], zero-padded rows (B operands / weights)
__global__ void cvt_w_kernel(const float* __restrict__ src, _Float16* __restrict__ dst,
                             int K, int Kp, int N) {
  long long total = (long long)Kp * N;
  long long i = (long long)blockIdx.x * blockDim.x + threadIdx.x;
  long long s = (long long)gridDim.x * blockDim.x;
  for (; i < total; i += s) {
    int r = (int)(i / N), c = (int)(i % N);
    dst[i] = (r < K) ? (_Float16)src[(long long)r * N + c] : (_Float16)0.f;
  }
}

// ---------------------------------------------------------------- WMMA GEMM
// C[M,N] = act(A[M,K] @ B[K,N] + bias[N]),  A/B are f16, C f32.
// REQUIRES: M % 16 == 0, K % 64 == 0, N % 64 == 0 (caller pads).
// 128 threads = 4 waves; block tile 16x64; K step 64 (2 wmma per iter).
// Fragments per CDNA5 ISA 7.12.2: A lane run is 2 contiguous 8-half chunks;
// B staged transposed in LDS so each lane reads the same 2 8-half chunks as A.
__global__ __launch_bounds__(128)
void gemm_wmma_kernel(const _Float16* __restrict__ A, const _Float16* __restrict__ B,
                      const float* __restrict__ bias, float* __restrict__ Cout,
                      int M, int K, int N, int act)
{
  __shared__ __align__(16) _Float16 As[16][64];    // 2 KB
  __shared__ __align__(16) _Float16 BsT[64][72];   // 9 KB, 144B row stride (16B aligned)

  const int tid  = threadIdx.x;
  const int lane = tid & 31;
  const int wid  = tid >> 5;              // 0..3 -> 16-wide N sub-tile
  const int m0   = blockIdx.x * 16;
  const int n0   = blockIdx.y * 64;

  const int mm  = lane & 15;
  const int nn  = (wid << 4) + (lane & 15);
  const int kg  = (lane >> 4) << 3;       // A: 0 or 8
  const int kg2 = (lane >> 4) << 4;       // unused (B reads use kg): 0 or 16

  v8f acc = {};

  for (int k0 = 0; k0 < K; k0 += 64) {
    if (k0 + 64 < K)
      __builtin_prefetch(&A[(long long)m0 * K + k0 + 64], 0, 1);

    // A tile 16x64 halves: 128 threads x one 16B (v8h) load
    {
      int r = tid >> 3, c8 = (tid & 7) << 3;
      *(v8h*)&As[r][c8] = *(const v8h*)&A[(long long)(m0 + r) * K + k0 + c8];
    }
    // B tile 64x64 halves: each thread loads 32 contiguous halves (4 x v8h),
    // stores transposed into BsT[n][k]
    {
      int r = tid >> 1, c32 = (tid & 1) << 5;
      const _Float16* gb = &B[(long long)(k0 + r) * N + n0 + c32];
      v8h t0 = *(const v8h*)(gb + 0);
      v8h t1 = *(const v8h*)(gb + 8);
      v8h t2 = *(const v8h*)(gb + 16);
      v8h t3 = *(const v8h*)(gb + 24);
#pragma unroll
      for (int j = 0; j < 8; ++j) {
        BsT[c32 + j     ][r] = t0[j];
        BsT[c32 + j + 8 ][r] = t1[j];
        BsT[c32 + j + 16][r] = t2[j];
        BsT[c32 + j + 24][r] = t3[j];
      }
    }
    __syncthreads();

#pragma unroll
    for (int c = 0; c < 2; ++c) {        // two 32-deep sub-chunks
      const int kb = c << 5;
      v8h alo = *(const v8h*)&As[mm][kb + kg];
      v8h ahi = *(const v8h*)&As[mm][kb + 16 + kg];
      v16h af = __builtin_shufflevector(alo, ahi,
                 0,1,2,3,4,5,6,7,8,9,10,11,12,13,14,15);
      v8h blo = *(const v8h*)&BsT[nn][kb + kg];
      v8h bhi = *(const v8h*)&BsT[nn][kb + 16 + kg];
      v16h bf = __builtin_shufflevector(blo, bhi,
                 0,1,2,3,4,5,6,7,8,9,10,11,12,13,14,15);
      acc = __builtin_amdgcn_wmma_f32_16x16x32_f16(
          false, af, false, bf, (short)0, acc, false, false);
    }
    __syncthreads();
  }

  // epilogue: C/D layout — VGPR j holds M = j + (lane>=16 ? 8 : 0), N = lane&15
  const int nc = n0 + nn;
  const float bv = bias ? bias[nc] : 0.f;
#pragma unroll
  for (int j = 0; j < 8; ++j) {
    int gm = m0 + j + ((lane >> 4) << 3);
    float v = acc[j] + bv;
    if (act == 1) v = fmaxf(v, 0.f);
    else if (act == 2) v = tanhf(v);
    Cout[(long long)gm * N + nc] = v;
  }
}

// ---------------------------------------------------------------- GCN pieces

__global__ void deg_kernel(const int* __restrict__ ei, float* __restrict__ deg) {
  int i = blockIdx.x * blockDim.x + threadIdx.x;
  int s = gridDim.x * blockDim.x;
  for (; i < NEDGES; i += s) atomicAdd(&deg[ei[NEDGES + i]], 1.0f);
}

__global__ void dinv_kernel(float* __restrict__ deg) {
  int i = blockIdx.x * blockDim.x + threadIdx.x;
  int s = gridDim.x * blockDim.x;
  for (; i < NNODES; i += s) deg[i] = rsqrtf(deg[i] + 1.0f);
}

__global__ void gcn_scatter_kernel(const int* __restrict__ ei, const float* __restrict__ h,
                                   const float* __restrict__ dinv, float* __restrict__ agg, int D) {
  long long total = (long long)NEDGES * D;
  long long i = (long long)blockIdx.x * blockDim.x + threadIdx.x;
  long long s = (long long)gridDim.x * blockDim.x;
  for (; i < total; i += s) {
    int e = (int)(i / D), d = (int)(i % D);
    int sn = ei[e], dn = ei[NEDGES + e];
    atomicAdd(&agg[(long long)dn * D + d], h[(long long)sn * D + d] * dinv[sn] * dinv[dn]);
  }
}

// h_relu = relu(agg + h*dinv^2 + b); dual write f32 + f16 (f16 feeds next WMMA GEMM)
__global__ void gcn_finish_kernel(const float* __restrict__ agg, const float* __restrict__ h,
                                  const float* __restrict__ dinv, const float* __restrict__ b,
                                  float* __restrict__ out, _Float16* __restrict__ outh, int D) {
  long long total = (long long)NNODES * D;
  long long i = (long long)blockIdx.x * blockDim.x + threadIdx.x;
  long long s = (long long)gridDim.x * blockDim.x;
  for (; i < total; i += s) {
    int n = (int)(i / D), d = (int)(i % D);
    float di = dinv[n];
    float v = fmaxf(agg[i] + h[i] * di * di + b[d], 0.f);
    out[i] = v;
    outh[i] = (_Float16)v;
  }
}

// ---------------------------------------------------------------- GAT pieces

__global__ void gat_alpha_kernel(const float* __restrict__ hg, const float* __restrict__ a_s,
                                 const float* __restrict__ a_d, float* __restrict__ asrc,
                                 float* __restrict__ adst, int D, int C) {
  int i = blockIdx.x * blockDim.x + threadIdx.x;
  int s = gridDim.x * blockDim.x;
  for (; i < NNODES * NHEADS; i += s) {
    int n = i >> 2, hh = i & 3;
    const float* base = hg + (long long)n * D + hh * C;
    float ss = 0.f, sd = 0.f;
    for (int c = 0; c < C; ++c) {
      float v = base[c];
      ss += v * a_s[hh * C + c];
      sd += v * a_d[hh * C + c];
    }
    asrc[i] = ss; adst[i] = sd;
  }
}

__global__ void gat_self_emax_kernel(const float* __restrict__ asrc, const float* __restrict__ adst,
                                     float* __restrict__ emax) {
  int i = blockIdx.x * blockDim.x + threadIdx.x;
  int s = gridDim.x * blockDim.x;
  for (; i < NNODES * NHEADS; i += s) emax[i] = leaky02(asrc[i] + adst[i]);
}

__global__ void gat_edge_max_kernel(const int* __restrict__ ei, const float* __restrict__ asrc,
                                    const float* __restrict__ adst, float* __restrict__ emax) {
  int i = blockIdx.x * blockDim.x + threadIdx.x;
  int s = gridDim.x * blockDim.x;
  for (; i < NEDGES * NHEADS; i += s) {
    int e = i >> 2, hh = i & 3;
    int sn = ei[e], dn = ei[NEDGES + e];
    atomicMaxF(&emax[dn * NHEADS + hh], leaky02(asrc[sn * NHEADS + hh] + adst[dn * NHEADS + hh]));
  }
}

__global__ void gat_self_sum_kernel(const float* __restrict__ asrc, const float* __restrict__ adst,
                                    const float* __restrict__ emax, float* __restrict__ esum) {
  int i = blockIdx.x * blockDim.x + threadIdx.x;
  int s = gridDim.x * blockDim.x;
  for (; i < NNODES * NHEADS; i += s)
    esum[i] = __expf(leaky02(asrc[i] + adst[i]) - emax[i]);
}

__global__ void gat_edge_sum_kernel(const int* __restrict__ ei, const float* __restrict__ asrc,
                                    const float* __restrict__ adst, const float* __restrict__ emax,
                                    float* __restrict__ esum) {
  int i = blockIdx.x * blockDim.x + threadIdx.x;
  int s = gridDim.x * blockDim.x;
  for (; i < NEDGES * NHEADS; i += s) {
    int e = i >> 2, hh = i & 3;
    int sn = ei[e], dn = ei[NEDGES + e];
    float v = __expf(leaky02(asrc[sn * NHEADS + hh] + adst[dn * NHEADS + hh]) - emax[dn * NHEADS + hh]);
    atomicAdd(&esum[dn * NHEADS + hh], v);
  }
}

__global__ void gat_edge_w_kernel(const int* __restrict__ ei, const float* __restrict__ asrc,
                                  const float* __restrict__ adst, const float* __restrict__ emax,
                                  const float* __restrict__ esum, float* __restrict__ walpha) {
  int i = blockIdx.x * blockDim.x + threadIdx.x;
  int s = gridDim.x * blockDim.x;
  for (; i < NEDGES * NHEADS; i += s) {
    int e = i >> 2, hh = i & 3;
    int sn = ei[e], dn = ei[NEDGES + e];
    float v = __expf(leaky02(asrc[sn * NHEADS + hh] + adst[dn * NHEADS + hh]) - emax[dn * NHEADS + hh]);
    walpha[i] = v / esum[dn * NHEADS + hh];
  }
}

__global__ void gat_self_w_kernel(const float* __restrict__ asrc, const float* __restrict__ adst,
                                  const float* __restrict__ emax, const float* __restrict__ esum,
                                  float* __restrict__ wself) {
  int i = blockIdx.x * blockDim.x + threadIdx.x;
  int s = gridDim.x * blockDim.x;
  for (; i < NNODES * NHEADS; i += s)
    wself[i] = __expf(leaky02(asrc[i] + adst[i]) - emax[i]) / esum[i];
}

__global__ void gat_scatter_kernel(const int* __restrict__ ei, const float* __restrict__ hg,
                                   const float* __restrict__ walpha, float* __restrict__ agg,
                                   int D, int C) {
  long long total = (long long)NEDGES * D;
  long long i = (long long)blockIdx.x * blockDim.x + threadIdx.x;
  long long s = (long long)gridDim.x * blockDim.x;
  for (; i < total; i += s) {
    int e = (int)(i / D), d = (int)(i % D);
    int sn = ei[e], dn = ei[NEDGES + e];
    atomicAdd(&agg[(long long)dn * D + d], hg[(long long)sn * D + d] * walpha[e * NHEADS + d / C]);
  }
}

// x_out = h_relu + agg + hg*wself + b; dual write f32 + f16
__global__ void gat_combine_kernel(const float* __restrict__ hr, const float* __restrict__ agg,
                                   const float* __restrict__ hg, const float* __restrict__ wself,
                                   const float* __restrict__ b, float* __restrict__ xout,
                                   _Float16* __restrict__ xouth, int D, int C) {
  long long total = (long long)NNODES * D;
  long long i = (long long)blockIdx.x * blockDim.x + threadIdx.x;
  long long s = (long long)gridDim.x * blockDim.x;
  for (; i < total; i += s) {
    int n = (int)(i / D), d = (int)(i % D);
    float v = hr[i] + agg[i] + hg[i] * wself[n * NHEADS + d / C] + b[d];
    xout[i] = v;
    xouth[i] = (_Float16)v;
  }
}

// ---------------------------------------------------------------- pooling

__global__ void pool_logit_kernel(const float* __restrict__ t, const float* __restrict__ w2,
                                  const float* __restrict__ b2, float* __restrict__ alog) {
  int i = blockIdx.x * blockDim.x + threadIdx.x;
  int s = gridDim.x * blockDim.x;
  for (; i < NNODES; i += s) {
    float acc = 0.f;
    const float* base = t + (long long)i * (DMAX / 2);
    for (int c = 0; c < DMAX / 2; ++c) acc += base[c] * w2[c];
    alog[i] = acc + b2[0];
  }
}

__global__ void pool_gmax_kernel(const float* __restrict__ alog, float* __restrict__ gmax) {
  int i = blockIdx.x * blockDim.x + threadIdx.x;
  int s = gridDim.x * blockDim.x;
  for (; i < NNODES; i += s) atomicMaxF(gmax, alog[i]);
}

__global__ void pool_exp_kernel(const float* __restrict__ alog, const float* __restrict__ gmax,
                                float* __restrict__ aexp, float* __restrict__ gsum) {
  int i = blockIdx.x * blockDim.x + threadIdx.x;
  int s = gridDim.x * blockDim.x;
  for (; i < NNODES; i += s) {
    float v = __expf(alog[i] - gmax[0]);
    aexp[i] = v;
    atomicAdd(gsum, v);
  }
}

__global__ void pool_seg_kernel(const int* __restrict__ batch, const float* __restrict__ aexp,
                                float* __restrict__ segexp, float* __restrict__ cnt) {
  int i = blockIdx.x * blockDim.x + threadIdx.x;
  int s = gridDim.x * blockDim.x;
  for (; i < NNODES; i += s) {
    int b = batch[i];
    atomicAdd(&segexp[b], aexp[i]);
    atomicAdd(&cnt[b], 1.0f);
  }
}

__global__ void out_init_kernel(float* __restrict__ out) {
  long long total = (long long)NGRAPH * OUTSTRIDE;
  long long i = (long long)blockIdx.x * blockDim.x + threadIdx.x;
  long long s = (long long)gridDim.x * blockDim.x;
  for (; i < total; i += s) {
    int c = (int)(i % OUTSTRIDE);
    out[i] = (c >= 1024 && c < 1536) ? -3.402823466e38f : 0.f;  // smax slab -> -FLT_MAX
  }
}

__global__ void pool_scatter_kernel(const int* __restrict__ batch, const float* __restrict__ x,
                                    const float* __restrict__ aexp, const float* __restrict__ segexp,
                                    const float* __restrict__ gsum, float* __restrict__ out) {
  long long total = (long long)NNODES * DMAX;
  long long i = (long long)blockIdx.x * blockDim.x + threadIdx.x;
  long long s = (long long)gridDim.x * blockDim.x;
  for (; i < total; i += s) {
    int n = (int)(i / DMAX), d = (int)(i % DMAX);
    int b = batch[n];
    float Z = gsum[0];
    float coeff = (aexp[n] / Z) / (segexp[b] / Z + 1e-8f);
    float xv = x[i];
    float* row = out + (long long)b * OUTSTRIDE;
    atomicAdd(&row[d], coeff * xv);           // graph_emb
    atomicMaxF(&row[1024 + d], xv);           // segment max
    atomicAdd(&row[1536 + d], xv);            // segment sum
  }
}

__global__ void pool_mean_kernel(const float* __restrict__ cnt, float* __restrict__ out) {
  long long total = (long long)NGRAPH * DMAX;
  long long i = (long long)blockIdx.x * blockDim.x + threadIdx.x;
  long long s = (long long)gridDim.x * blockDim.x;
  for (; i < total; i += s) {
    int b = (int)(i / DMAX), d = (int)(i % DMAX);
    float* row = out + (long long)b * OUTSTRIDE;
    row[512 + d] = row[1536 + d] / fmaxf(cnt[b], 1.0f);
  }
}

// ---------------------------------------------------------------- launcher

static inline int nblk(long long total) {
  long long b = (total + 255) / 256;
  if (b > 262144) b = 262144;
  if (b < 1) b = 1;
  return (int)b;
}

extern "C" void kernel_launch(void* const* d_in, const int* in_sizes, int n_in,
                              void* d_out, int out_size, void* d_ws, size_t ws_size,
                              hipStream_t stream) {
  const float* x0    = (const float*)d_in[0];
  const int*   ei    = (const int*)d_in[1];
  const int*   batch = (const int*)d_in[2];
  const float* ap_w1 = (const float*)d_in[21];
  const float* ap_b1 = (const float*)d_in[22];
  const float* ap_w2 = (const float*)d_in[23];
  const float* ap_b2 = (const float*)d_in[24];
  float* out = (float*)d_out;

  // workspace carve-up
  float* ws = (float*)d_ws;
  const long long ND = (long long)NNODES * DMAX;
  float* X0   = ws;               ws += ND;
  float* X1   = ws;               ws += ND;
  float* HR   = ws;               ws += ND;        // relu(GCN) output (f32)
  float* HG   = ws;               ws += ND;        // GEMM outputs (f32)
  float* AGG  = ws;               ws += ND;        // scatter accumulator
  float* dinv = ws;               ws += NNODES;
  float* asrc = ws;               ws += NNODES * NHEADS;
  float* adst = ws;               ws += NNODES * NHEADS;
  float* emax = ws;               ws += NNODES * NHEADS;
  float* esum = ws;               ws += NNODES * NHEADS;
  float* wslf = ws;               ws += NNODES * NHEADS;
  float* walp = ws;               ws += (long long)NEDGES * NHEADS;
  float* alog = ws;               ws += NNODES;
  float* aexp = ws;               ws += NNODES;
  float* gmax = ws;               ws += 1;
  float* gsum = ws;               ws += 1;
  float* cnt  = ws;               ws += NGRAPH;
  float* segx = ws;               ws += NGRAPH + 2; // +pad for f16 alignment
  _Float16* XH  = (_Float16*)ws;  ws += ND / 2;     // f16 GEMM A-operand (x / x_out)
  _Float16* HRH = (_Float16*)ws;  ws += ND / 2;     // f16 GEMM A-operand (h_relu)
  _Float16* WH  = (_Float16*)ws;  ws += (512 * 512) / 2;  // f16 weight staging

  // ---- degree normalization (edge-only, shared across layers)
  fill_kernel<<<nblk(NNODES), 256, 0, stream>>>(dinv, 0.f, NNODES);
  deg_kernel<<<nblk(NEDGES), 256, 0, stream>>>(ei, dinv);
  dinv_kernel<<<nblk(NNODES), 256, 0, stream>>>(dinv);

  const int douts[3] = {128, 256, 512};
  const int dins[3]  = {36, 128, 256};
  const int dinps[3] = {64, 128, 256};   // K padded to multiple of 64
  float* xswap[3] = {X0, X1, X0};

  // layer-0 input: f32 [10000 x 36] -> f16 [10000 x 64] zero-padded
  cvt_a_kernel<<<nblk((long long)NNODES * 64), 256, 0, stream>>>(x0, XH, NNODES, 36, 64);

  for (int l = 0; l < 3; ++l) {
    const float* gcn_w  = (const float*)d_in[3 + 6 * l + 0];
    const float* gcn_b  = (const float*)d_in[3 + 6 * l + 1];
    const float* gat_w  = (const float*)d_in[3 + 6 * l + 2];
    const float* gat_as = (const float*)d_in[3 + 6 * l + 3];
    const float* gat_ad = (const float*)d_in[3 + 6 * l + 4];
    const float* gat_b  = (const float*)d_in[3 + 6 * l + 5];
    const int din = dins[l], dinp = dinps[l], dout = douts[l], C = dout / NHEADS;
    const long long nd = (long long)NNODES * dout;
    dim3 ggrid(NNODES / 16 + (NNODES % 16 != 0), dout / 64);

    // GCN: h_lin = x @ gcn_w  (WMMA, into HG)
    cvt_w_kernel<<<nblk((long long)dinp * dout), 256, 0, stream>>>(gcn_w, WH, din, dinp, dout);
    gemm_wmma_kernel<<<ggrid, 128, 0, stream>>>(XH, WH, nullptr, HG, NNODES, dinp, dout, 0);
    fill_kernel<<<nblk(nd), 256, 0, stream>>>(AGG, 0.f, nd);
    gcn_scatter_kernel<<<nblk((long long)NEDGES * dout), 256, 0, stream>>>(ei, HG, dinv, AGG, dout);
    gcn_finish_kernel<<<nblk(nd), 256, 0, stream>>>(AGG, HG, dinv, gcn_b, HR, HRH, dout);

    // GAT: hg = h_relu @ gat_w  (WMMA, into HG)
    cvt_w_kernel<<<nblk((long long)dout * dout), 256, 0, stream>>>(gat_w, WH, dout, dout, dout);
    gemm_wmma_kernel<<<ggrid, 128, 0, stream>>>(HRH, WH, nullptr, HG, NNODES, dout, dout, 0);
    gat_alpha_kernel<<<nblk(NNODES * NHEADS), 256, 0, stream>>>(HG, gat_as, gat_ad, asrc, adst, dout, C);
    gat_self_emax_kernel<<<nblk(NNODES * NHEADS), 256, 0, stream>>>(asrc, adst, emax);
    gat_edge_max_kernel<<<nblk((long long)NEDGES * NHEADS), 256, 0, stream>>>(ei, asrc, adst, emax);
    gat_self_sum_kernel<<<nblk(NNODES * NHEADS), 256, 0, stream>>>(asrc, adst, emax, esum);
    gat_edge_sum_kernel<<<nblk((long long)NEDGES * NHEADS), 256, 0, stream>>>(ei, asrc, adst, emax, esum);
    gat_edge_w_kernel<<<nblk((long long)NEDGES * NHEADS), 256, 0, stream>>>(ei, asrc, adst, emax, esum, walp);
    gat_self_w_kernel<<<nblk(NNODES * NHEADS), 256, 0, stream>>>(asrc, adst, emax, esum, wslf);
    fill_kernel<<<nblk(nd), 256, 0, stream>>>(AGG, 0.f, nd);
    gat_scatter_kernel<<<nblk((long long)NEDGES * dout), 256, 0, stream>>>(ei, HG, walp, AGG, dout, C);
    gat_combine_kernel<<<nblk(nd), 256, 0, stream>>>(HR, AGG, HG, wslf, gat_b, xswap[l], XH, dout, C);
  }

  // ---- attention pooling + segment stats (final x: f32 in X0, f16 in XH, D=512)
  const float* xf = xswap[2];
  dim3 pgrid(NNODES / 16 + (NNODES % 16 != 0), (DMAX / 2) / 64);
  cvt_w_kernel<<<nblk((long long)DMAX * (DMAX / 2)), 256, 0, stream>>>(ap_w1, WH, DMAX, DMAX, DMAX / 2);
  gemm_wmma_kernel<<<pgrid, 128, 0, stream>>>(XH, WH, ap_b1, HR, NNODES, DMAX, DMAX / 2, 2);
  pool_logit_kernel<<<nblk(NNODES), 256, 0, stream>>>(HR, ap_w2, ap_b2, alog);

  fill_kernel<<<1, 64, 0, stream>>>(gmax, -3.402823466e38f, 1);
  fill_kernel<<<1, 64, 0, stream>>>(gsum, 0.f, 1);
  fill_kernel<<<1, 256, 0, stream>>>(cnt, 0.f, NGRAPH);
  fill_kernel<<<1, 256, 0, stream>>>(segx, 0.f, NGRAPH);

  pool_gmax_kernel<<<nblk(NNODES), 256, 0, stream>>>(alog, gmax);
  pool_exp_kernel<<<nblk(NNODES), 256, 0, stream>>>(alog, gmax, aexp, gsum);
  pool_seg_kernel<<<nblk(NNODES), 256, 0, stream>>>(batch, aexp, segx, cnt);

  out_init_kernel<<<nblk((long long)NGRAPH * OUTSTRIDE), 256, 0, stream>>>(out);
  pool_scatter_kernel<<<nblk((long long)NNODES * DMAX), 256, 0, stream>>>(batch, xf, aexp, segx, gsum, out);
  pool_mean_kernel<<<nblk((long long)NGRAPH * DMAX), 256, 0, stream>>>(cnt, out);
}
